// GNNsimple_75368086110725
// MI455X (gfx1250) — matrix operand order repaired and checked by reading, not compiled, not run on hardware
//
#include <hip/hip_runtime.h>
#include <hip/hip_bf16.h>

#define FDIM 128
#define NUM_GRAPHS 64
#define MTILE 64              // output rows per block in the WMMA GEMM

typedef __attribute__((ext_vector_type(16))) _Float16 v16h;
typedef __attribute__((ext_vector_type(4)))  _Float16 v4h;
typedef __attribute__((ext_vector_type(8)))  float    v8f;

// CDNA5 async global->LDS copy (16B per lane), tracked by ASYNCcnt.
// lds_addr = low 32 bits of generic shared pointer (== wave LDS byte offset).
__device__ __forceinline__ void async_g2lds_b128(unsigned int lds_addr, unsigned long long gaddr) {
    asm volatile("global_load_async_to_lds_b128 %0, %1, off"
                 :: "v"(lds_addr), "v"(gaddr) : "memory");
}
__device__ __forceinline__ void wait_asynccnt0() {
    asm volatile("s_wait_asynccnt 0x0" ::: "memory");
}

// ---------------------------------------------------------------------------
// deg[dst] += 1 per edge
__global__ __launch_bounds__(256)
void degree_kernel(const int* __restrict__ ei, int E, float* __restrict__ deg) {
    int t = blockIdx.x * 256 + threadIdx.x;
    if (t < E) atomicAdd(&deg[ei[E + t]], 1.0f);
}

// invdeg[i] = 1 / max(deg,1)
__global__ __launch_bounds__(256)
void invdeg_kernel(const float* __restrict__ deg, float* __restrict__ ivd, int N) {
    int t = blockIdx.x * 256 + threadIdx.x;
    if (t < N) ivd[t] = 1.0f / fmaxf(deg[t], 1.0f);
}

// f32 -> f16 conversion, 4 elements/thread (n must be a multiple of 4)
__global__ __launch_bounds__(256)
void cvt_f16(const float* __restrict__ src, _Float16* __restrict__ dst, int n) {
    int t = blockIdx.x * 256 + threadIdx.x;
    int i = t * 4;
    if (i < n) {
        float4 v = ((const float4*)src)[t];
        v4h h; h[0] = (_Float16)v.x; h[1] = (_Float16)v.y;
               h[2] = (_Float16)v.z; h[3] = (_Float16)v.w;
        *(v4h*)(dst + i) = h;
    }
}

// one wave per edge: dstsum[dst,:] += feat[src,:] in f16
// lane handles 4 features = 8B gather + 2x global_atomic_pk_add_f16
__global__ __launch_bounds__(256)
void edge_scatter_f16(const int* __restrict__ ei, int E,
                      const _Float16* __restrict__ feat, _Float16* __restrict__ dstsum) {
    long long t = (long long)blockIdx.x * 256 + threadIdx.x;
    int e    = (int)(t >> 5);
    int lane = (int)(t & 31);
    if (e >= E) return;
    int s = ei[e];
    int d = ei[E + e];
    const uint2 v = ((const uint2*)(feat + (size_t)s * FDIM))[lane];   // 4 halves
    unsigned long long addr = (unsigned long long)(dstsum + (size_t)d * FDIM + lane * 4);
    asm volatile("global_atomic_pk_add_f16 %0, %1, off\n\t"
                 "global_atomic_pk_add_f16 %0, %2, off offset:4"
                 :: "v"(addr), "v"(v.x), "v"(v.y) : "memory");
}

// one wave per node: agg[i,:] *= invdeg[i]   (f16, 4 halves per lane)
__global__ __launch_bounds__(256)
void norm_agg(_Float16* __restrict__ agg, const float* __restrict__ ivd, int N) {
    long long t = (long long)blockIdx.x * 256 + threadIdx.x;
    int i    = (int)(t >> 5);
    int lane = (int)(t & 31);
    if (i >= N) return;
    _Float16 s = (_Float16)ivd[i];
    v4h* p = (v4h*)(agg + (size_t)i * FDIM) + lane;
    v4h v = *p;
    v = v * s;
    *p = v;
}

// ---------------------------------------------------------------------------
// Fused SAGE layer: out[i,:] = relu( aggn[i,:] @ Wl^T + bl + self[i,:] @ Wr^T )
// aggn/self are f16; A tiles staged via async global->LDS copies.
// Block = 64 rows (4 sub-tiles of 16), 8 waves; wave w owns cols [16w,16w+16).
template <int OUTF16>
__global__ __launch_bounds__(256)
void sage_gemm(const _Float16* __restrict__ aggn, const _Float16* __restrict__ self,
               const _Float16* __restrict__ Wl16, const float* __restrict__ bl,
               const _Float16* __restrict__ Wr16,
               void* __restrict__ outp, int N) {
    __shared__ _Float16 sA[MTILE][FDIM];   // normalized aggregate tile
    __shared__ _Float16 sX[MTILE][FDIM];   // self-features tile

    const int tid = threadIdx.x;
    const int m0  = blockIdx.x * MTILE;

    // Async stage: 64 rows x 256B per matrix; each lane copies 16B chunks.
    const unsigned int ldsA0 = (unsigned int)(size_t)&sA[0][0];
    const unsigned int ldsX0 = (unsigned int)(size_t)&sX[0][0];
    for (int c = tid; c < MTILE * 16; c += 256) {       // 4 iterations, all lanes live
        int r   = c >> 4;
        int seg = c & 15;
        int row = m0 + r;
        unsigned int boff = (unsigned int)(r * (FDIM * 2) + seg * 16);
        if (row < N) {
            unsigned long long ga = (unsigned long long)((const char*)(aggn + (size_t)row * FDIM) + seg * 16);
            unsigned long long gx = (unsigned long long)((const char*)(self + (size_t)row * FDIM) + seg * 16);
            async_g2lds_b128(ldsA0 + boff, ga);
            async_g2lds_b128(ldsX0 + boff, gx);
        } else {
            uint4 z = {0u, 0u, 0u, 0u};
            *(uint4*)((char*)&sA[0][0] + boff) = z;
            *(uint4*)((char*)&sX[0][0] + boff) = z;
        }
    }
    wait_asynccnt0();
    __syncthreads();

    const int lane = tid & 31;
    const int wave = tid >> 5;
    const int n0   = wave * 16;
    const int lm   = lane & 15;   // A: row in tile; B: column in tile
    const int lh   = lane >> 4;   // half-wave select

    // B fragments: lane holds col n0+lm; K = kb + lh*8 + j (j<8), kb + 16 + lh*8 + (j-8) (j>=8); B[k][n] = W[n][k].
    v16h bf[8];
    #pragma unroll
    for (int f = 0; f < 8; ++f) {
        const _Float16* W16 = (f < 4) ? Wl16 : Wr16;
        const int kb = (f & 3) * 32;
        const _Float16* wp = W16 + (size_t)(n0 + lm) * FDIM + kb + lh * 8;
        #pragma unroll
        for (int j = 0; j < 16; ++j) bf[f][j] = wp[(j < 8) ? j : (8 + j)];
    }

    v8f acc[MTILE / 16] = {};
    #pragma unroll
    for (int f = 0; f < 8; ++f) {
        const int kb = (f & 3) * 32;
        #pragma unroll
        for (int mt = 0; mt < MTILE / 16; ++mt) {
            // A fragment (16x32 f16): lanes 0-15: K kb+0..7/kb+16..23; lanes 16-31: +8
            v16h a;
            const _Float16* ap = (f < 4) ? &sA[mt * 16 + lm][kb + lh * 8]
                                         : &sX[mt * 16 + lm][kb + lh * 8];
            #pragma unroll
            for (int j = 0; j < 8; ++j) a[j] = ap[j];
            #pragma unroll
            for (int j = 0; j < 8; ++j) a[8 + j] = ap[16 + j];
            acc[mt] = __builtin_amdgcn_wmma_f32_16x16x32_f16(
                          false, a, false, bf[f], (short)0, acc[mt], false, false);
        }
    }

    // Epilogue: lane holds col n0+lm; VGPR r -> row mt*16 + r + lh*8.
    const int n = n0 + lm;
    const float bias = bl[n];
    #pragma unroll
    for (int mt = 0; mt < MTILE / 16; ++mt) {
        #pragma unroll
        for (int r = 0; r < 8; ++r) {
            int row = m0 + mt * 16 + r + lh * 8;
            if (row < N) {
                float v = fmaxf(acc[mt][r] + bias, 0.0f);
                if (OUTF16) ((_Float16*)outp)[(size_t)row * FDIM + n] = (_Float16)v;
                else        ((float*)outp)[(size_t)row * FDIM + n] = v;
            }
        }
    }
}

// ---------------------------------------------------------------------------
// one wave per node: gsum[batch[i],:] += h[i,:]; gcnt[batch[i]] += 1  (f32)
__global__ __launch_bounds__(256)
void pool_scatter(const float* __restrict__ h, const int* __restrict__ batch, int N,
                  float* __restrict__ gsum, float* __restrict__ gcnt) {
    long long t = (long long)blockIdx.x * 256 + threadIdx.x;
    int i    = (int)(t >> 5);
    int lane = (int)(t & 31);
    if (i >= N) return;
    int g = batch[i];
    const float4 v = ((const float4*)(h + (size_t)i * FDIM))[lane];
    float* o = gsum + (size_t)g * FDIM + lane * 4;
    atomicAdd(o + 0, v.x); atomicAdd(o + 1, v.y);
    atomicAdd(o + 2, v.z); atomicAdd(o + 3, v.w);
    if (lane == 0) atomicAdd(&gcnt[g], 1.0f);
}

// one block per graph: gemb = gsum/cnt; z = relu(gemb@Wc1^T+bc1); out = z@Wc2^T+bc2
__global__ __launch_bounds__(128)
void classifier(const float* __restrict__ gsum, const float* __restrict__ gcnt,
                const float* __restrict__ Wc1, const float* __restrict__ bc1,
                const float* __restrict__ Wc2, const float* __restrict__ bc2,
                float* __restrict__ out) {
    __shared__ float ge[FDIM];
    __shared__ float z[FDIM];
    const int g = blockIdx.x;
    const int j = threadIdx.x;
    float inv = 1.0f / fmaxf(gcnt[g], 1.0f);
    ge[j] = gsum[(size_t)g * FDIM + j] * inv;
    __syncthreads();
    float acc = bc1[j];
    const float* wr = Wc1 + (size_t)j * FDIM;
    #pragma unroll 8
    for (int k = 0; k < FDIM; ++k) acc += ge[k] * wr[k];
    z[j] = fmaxf(acc, 0.0f) * Wc2[j];
    __syncthreads();
    for (int s = FDIM / 2; s > 0; s >>= 1) {
        if (j < s) z[j] += z[j + s];
        __syncthreads();
    }
    if (j == 0) out[g] = z[0] + bc2[0];
}

// ---------------------------------------------------------------------------
extern "C" void kernel_launch(void* const* d_in, const int* in_sizes, int n_in,
                              void* d_out, int out_size, void* d_ws, size_t ws_size,
                              hipStream_t stream) {
    const float* x     = (const float*)d_in[0];
    const int*   ei    = (const int*)d_in[1];
    const int*   batch = (const int*)d_in[2];
    const float* W1l   = (const float*)d_in[3];
    const float* b1l   = (const float*)d_in[4];
    const float* W1r   = (const float*)d_in[5];
    const float* W2l   = (const float*)d_in[6];
    const float* b2l   = (const float*)d_in[7];
    const float* W2r   = (const float*)d_in[8];
    const float* Wc1   = (const float*)d_in[9];
    const float* bc1   = (const float*)d_in[10];
    const float* Wc2   = (const float*)d_in[11];
    const float* bc2   = (const float*)d_in[12];
    float* out = (float*)d_out;

    const int N  = in_sizes[0] / FDIM;
    const int E  = in_sizes[1] / 2;
    const int WN = FDIM * FDIM;                    // 16384 elems per weight matrix
    const int NF = N * FDIM;

    // workspace layout (float units, every region 16B aligned)
    float* ws = (float*)d_ws;
    size_t off = 0;
    auto align4 = [](size_t v) { return (v + 3) & ~(size_t)3; };
    float*    deg   = ws + off; off = align4(off + N);
    float*    ivd   = ws + off; off = align4(off + N);
    _Float16* agg16 = (_Float16*)(ws + off); off = align4(off + (size_t)NF / 2);
    _Float16* x16   = (_Float16*)(ws + off); off = align4(off + (size_t)NF / 2);
    _Float16* h116  = (_Float16*)(ws + off); off = align4(off + (size_t)NF / 2);
    float*    h2    = ws + off; off = align4(off + (size_t)NF);
    float*    gsum  = ws + off; off = align4(off + NUM_GRAPHS * FDIM);
    float*    gcnt  = ws + off; off = align4(off + NUM_GRAPHS);
    _Float16* w16   = (_Float16*)(ws + off);
    _Float16* W1l16 = w16;
    _Float16* W1r16 = w16 + WN;
    _Float16* W2l16 = w16 + 2 * WN;
    _Float16* W2r16 = w16 + 3 * WN;

    hipMemsetAsync(deg,   0, sizeof(float) * N, stream);
    hipMemsetAsync(agg16, 0, sizeof(_Float16) * (size_t)NF, stream);
    hipMemsetAsync(gsum,  0, sizeof(float) * (NUM_GRAPHS * FDIM + NUM_GRAPHS + 4), stream);

    const int edgeWaveBlocks = (int)(((long long)E * 32 + 255) / 256);
    const int nodeWaveBlocks = (int)(((long long)N * 32 + 255) / 256);
    const int wBlocks = (WN / 4 + 255) / 256;
    const int xBlocks = (NF / 4 + 255) / 256;

    // one-time (per call) f16 conversions
    cvt_f16<<<wBlocks, 256, 0, stream>>>(W1l, W1l16, WN);
    cvt_f16<<<wBlocks, 256, 0, stream>>>(W1r, W1r16, WN);
    cvt_f16<<<wBlocks, 256, 0, stream>>>(W2l, W2l16, WN);
    cvt_f16<<<wBlocks, 256, 0, stream>>>(W2r, W2r16, WN);
    cvt_f16<<<xBlocks, 256, 0, stream>>>(x, x16, NF);

    degree_kernel<<<(E + 255) / 256, 256, 0, stream>>>(ei, E, deg);
    invdeg_kernel<<<(N + 255) / 256, 256, 0, stream>>>(deg, ivd, N);

    const int gemmBlocks = (N + MTILE - 1) / MTILE;

    // Layer 1
    edge_scatter_f16<<<edgeWaveBlocks, 256, 0, stream>>>(ei, E, x16, agg16);
    norm_agg<<<nodeWaveBlocks, 256, 0, stream>>>(agg16, ivd, N);
    sage_gemm<1><<<gemmBlocks, 256, 0, stream>>>(agg16, x16, W1l16, b1l, W1r16, (void*)h116, N);

    // Layer 2
    hipMemsetAsync(agg16, 0, sizeof(_Float16) * (size_t)NF, stream);
    edge_scatter_f16<<<edgeWaveBlocks, 256, 0, stream>>>(ei, E, h116, agg16);
    norm_agg<<<nodeWaveBlocks, 256, 0, stream>>>(agg16, ivd, N);
    sage_gemm<0><<<gemmBlocks, 256, 0, stream>>>(agg16, h116, W2l16, b2l, W2r16, (void*)h2, N);

    // Pool + classify
    pool_scatter<<<nodeWaveBlocks, 256, 0, stream>>>(h2, batch, N, gsum, gcnt);
    classifier<<<NUM_GRAPHS, FDIM, 0, stream>>>(gsum, gcnt, Wc1, bc1, Wc2, bc2, out);
}
